// SimpleBEVDecoderLayer_pixel_20916490731829
// MI455X (gfx1250) — matrix operand order repaired and checked by reading, not compiled, not run on hardware
//
#include <hip/hip_runtime.h>

typedef __attribute__((ext_vector_type(16))) __bf16 v16bf;
typedef __attribute__((ext_vector_type(8)))  __bf16 v8bf;
typedef __attribute__((ext_vector_type(8)))  float  v8f;

#define BH 128
#define BW 128
#define BM (BH*BW)

// ---------- repack OIHW f32 -> WMMA-fragment-major bf16 ----------
// Logical B is [K][N], K = tap*I + i, N = O.  Stored so that for k-block kb
// (32 k's) and n-tile nb (16 cols), lane l's 16 bf16 fragment elements are
// contiguous:  dst[(((kb*(N/16)+nb)*32)+l)*16 + (kr&15)],  l = nr + (kr&16).
__global__ void repack_kernel(const float* __restrict__ src, __bf16* __restrict__ dst,
                              int O, int I, int R) {
    int idx = blockIdx.x * blockDim.x + threadIdx.x;
    int total = O * I * R;
    if (idx >= total) return;
    int n = idx % O;
    int t = idx / O;
    int i = t % I;
    int r = t / I;
    int k  = r * I + i;
    int kb = k >> 5, kr = k & 31;
    int nb = n >> 4, nr = n & 15;
    int lanev = nr + (kr & 16);
    size_t doff = ((((size_t)kb * (O >> 4) + nb) * 32) + lanev) * 16 + (kr & 15);
    dst[doff] = (__bf16)src[((size_t)n * I + i) * R + r];
}

// ---------- sampling_4d + distance weight + sum over P ----------
__global__ void sample_kernel(const float* __restrict__ feats,
                              const float* __restrict__ l2i,
                              const float* __restrict__ bev,
                              __bf16* __restrict__ x0) {
    __shared__ float sM[96];
    __shared__ float sPos[3];
    const int tid = threadIdx.x;
    if (tid < 96) sM[tid] = l2i[tid];
    __syncthreads();

    const int q = blockIdx.x;
    const int h = q >> 7, w = q & 127;
    const int c = tid;
    const bool isFeat = (c < 128);
    const bool isPos  = (c >= 128 && c < 131);
    const bool active = (c < 131);

    float acc = 0.f;
    #pragma unroll
    for (int p = 0; p < 4; ++p) {
        const size_t bi = (((size_t)p * BH + h) * BW + w) * 3;
        const float rx = bev[bi + 0] * 102.4f - 51.2f;
        const float ry = bev[bi + 1] * 102.4f - 51.2f;
        const float rz = (float)p * (1.0f / 3.0f) * 8.0f - 5.0f;
        float s = 0.f;
        if (active) {
            for (int n = 0; n < 6; ++n) {
                const float* Mm = sM + n * 16;
                const float cx = Mm[0]*rx + Mm[1]*ry + Mm[2] *rz + Mm[3];
                const float cy = Mm[4]*rx + Mm[5]*ry + Mm[6] *rz + Mm[7];
                const float cz = Mm[8]*rx + Mm[9]*ry + Mm[10]*rz + Mm[11];
                const float valid = (cz > 1e-5f) ? 1.f : 0.f;
                const float zc = fmaxf(cz, 1e-5f);
                const float px = cx / zc * (1.0f/704.0f) * 44.f - 0.5f;
                const float py = cy / zc * (1.0f/256.0f) * 16.f - 0.5f;
                const float xf = floorf(px), yf = floorf(py);
                const float wx = px - xf, wy = py - yf;
                const int ix = (int)xf, iy = (int)yf;
                const float* fm = feats + ((size_t)n * 131 + c) * (16 * 44);
                float b = 0.f;
                { int xi=ix,   yi=iy;   if (xi>=0&&xi<44&&yi>=0&&yi<16) b += fm[yi*44+xi]*(1.f-wx)*(1.f-wy); }
                { int xi=ix+1, yi=iy;   if (xi>=0&&xi<44&&yi>=0&&yi<16) b += fm[yi*44+xi]*wx*(1.f-wy); }
                { int xi=ix,   yi=iy+1; if (xi>=0&&xi<44&&yi>=0&&yi<16) b += fm[yi*44+xi]*(1.f-wx)*wy; }
                { int xi=ix+1, yi=iy+1; if (xi>=0&&xi<44&&yi>=0&&yi<16) b += fm[yi*44+xi]*wx*wy; }
                s += b * valid;
            }
        }
        if (isPos) sPos[c - 128] = s;
        __syncthreads();
        if (isFeat) {
            const float dx = rx - sPos[0], dy = ry - sPos[1], dz = rz - sPos[2];
            acc += s * __expf(-0.075f * (dx*dx + dy*dy + dz*dz));
        }
        __syncthreads();
    }
    if (isFeat) x0[(size_t)q * 128 + c] = (__bf16)acc;
}

// ---------- implicit-GEMM conv via v_wmma_f32_16x16x32_bf16 ----------
// A: [BM][CIN] bf16 (spatial-major).  Wt: fragment-major swizzled weights.
// MODE: 0 = f32 [m][N]; 1 = GELU -> bf16 [m][N]; 2 = f32 transposed [n][BM]
#define SA_STRIDE 40   // bf16 units; keeps every 8-elem chunk 16B aligned

template <int CIN, int CSHIFT, int TAPS, int NOUT, int MODE>
__global__ void __launch_bounds__(256)
gemm_conv_wmma(const __bf16* __restrict__ A, const __bf16* __restrict__ Wt,
               float* __restrict__ outF, __bf16* __restrict__ outB) {
    constexpr int K      = CIN * TAPS;
    constexpr int NSTEPS = K >> 5;          // even for all three convs
    constexpr int NTILES = NOUT >> 4;

    __shared__ __align__(16) __bf16 sA[2][64 * SA_STRIDE];

    const int m0 = blockIdx.x * 64;
    const int n0 = blockIdx.y * 128;
    const int tid  = threadIdx.x;
    const int lane = tid & 31;
    const int wave = tid >> 5;
    const int wm = (wave >> 2) * 32;   // 0 / 32
    const int wn = (wave & 3) * 32;    // 0,32,64,96

    v8f acc00 = {}, acc01 = {}, acc10 = {}, acc11 = {};

    // A staging: 64 rows x 32 k; each thread: one 8-bf16 chunk (never crosses a tap)
    const int ar = tid >> 2;
    const int ak = (tid & 3) * 8;
    const int ah = (m0 + ar) >> 7;
    const int aw = (m0 + ar) & 127;

    // fragment indices (contiguous k order: lane half h holds k = 16h..16h+15, as in Wt)
    const int arow = wm + (lane & 15);
    const int akb  = (lane >> 4) << 4;           // 0 or 16
    const int nt0  = (n0 + wn) >> 4;             // n-tile of b0; b1 = nt0+1

    auto stageA = [&](int step, __bf16* buf) {
        const int k   = (step << 5) + ak;
        const int tap = k >> CSHIFT;
        const int ci  = k & (CIN - 1);
        int dy = 0, dx = 0;
        if (TAPS != 1) { dy = tap / 3 - 1; dx = tap % 3 - 1; }
        const int hh = ah + dy, ww = aw + dx;
        uint4 val = make_uint4(0u, 0u, 0u, 0u);
        if (hh >= 0 && hh < BH && ww >= 0 && ww < BW)
            val = *reinterpret_cast<const uint4*>(A + ((size_t)hh * BW + ww) * CIN + ci);
        *reinterpret_cast<uint4*>(&buf[ar * SA_STRIDE + ak]) = val;
    };

    auto compute = [&](int step, const __bf16* buf) {
        // A fragments: four ds_load_b128
        const v8bf a0l = *reinterpret_cast<const v8bf*>(&buf[arow * SA_STRIDE + akb]);
        const v8bf a0h = *reinterpret_cast<const v8bf*>(&buf[arow * SA_STRIDE + akb + 8]);
        const v8bf a1l = *reinterpret_cast<const v8bf*>(&buf[(arow + 16) * SA_STRIDE + akb]);
        const v8bf a1h = *reinterpret_cast<const v8bf*>(&buf[(arow + 16) * SA_STRIDE + akb + 8]);
        const v16bf a0 = __builtin_shufflevector(a0l, a0h, 0,1,2,3,4,5,6,7,8,9,10,11,12,13,14,15);
        const v16bf a1 = __builtin_shufflevector(a1l, a1h, 0,1,2,3,4,5,6,7,8,9,10,11,12,13,14,15);

        // B fragments: coalesced 32B/lane straight from L2-resident weights
        const __bf16* b0p = Wt + ((((size_t)step * NTILES + nt0)     * 32) + lane) * 16;
        const __bf16* b1p = Wt + ((((size_t)step * NTILES + nt0 + 1) * 32) + lane) * 16;
        const v8bf b0l = *reinterpret_cast<const v8bf*>(b0p);
        const v8bf b0h = *reinterpret_cast<const v8bf*>(b0p + 8);
        const v8bf b1l = *reinterpret_cast<const v8bf*>(b1p);
        const v8bf b1h = *reinterpret_cast<const v8bf*>(b1p + 8);
        const v16bf b0 = __builtin_shufflevector(b0l, b0h, 0,1,2,3,4,5,6,7,8,9,10,11,12,13,14,15);
        const v16bf b1 = __builtin_shufflevector(b1l, b1h, 0,1,2,3,4,5,6,7,8,9,10,11,12,13,14,15);

        // prefetch next k-block's weight chunk (global_prefetch_b8)
        __builtin_prefetch(b0p + (size_t)NTILES * 32 * 16, 0, 1);

        acc00 = __builtin_amdgcn_wmma_f32_16x16x32_bf16(false, a0, false, b0, (short)0, acc00, false, false);
        acc01 = __builtin_amdgcn_wmma_f32_16x16x32_bf16(false, a0, false, b1, (short)0, acc01, false, false);
        acc10 = __builtin_amdgcn_wmma_f32_16x16x32_bf16(false, a1, false, b0, (short)0, acc10, false, false);
        acc11 = __builtin_amdgcn_wmma_f32_16x16x32_bf16(false, a1, false, b1, (short)0, acc11, false, false);
    };

    // software pipeline, double-buffered LDS, one barrier per K-step
    stageA(0, sA[0]);
    for (int s = 0; s < NSTEPS; s += 2) {
        __syncthreads();
        if (s + 1 < NSTEPS) stageA(s + 1, sA[1]);
        compute(s, sA[0]);
        __syncthreads();
        if (s + 2 < NSTEPS) stageA(s + 2, sA[0]);
        if (s + 1 < NSTEPS) compute(s + 1, sA[1]);
    }

    // epilogue: C/D layout M = vgpr + 8*(lane>=16), N = lane&15
    const int rbase = 8 * (lane >> 4);
    const int ncol  = lane & 15;
    #pragma unroll
    for (int fi = 0; fi < 2; ++fi) {
        #pragma unroll
        for (int fj = 0; fj < 2; ++fj) {
            v8f acc = (fi == 0) ? (fj == 0 ? acc00 : acc01)
                                : (fj == 0 ? acc10 : acc11);
            #pragma unroll
            for (int r = 0; r < 8; ++r) {
                const int mrow = m0 + wm + fi * 16 + rbase + r;
                const int nc   = n0 + wn + fj * 16 + ncol;
                const float v  = acc[r];
                if (MODE == 0) {
                    outF[(size_t)mrow * NOUT + nc] = v;
                } else if (MODE == 1) {
                    const float g = 0.5f * v * (1.0f + erff(v * 0.70710678118654752f));
                    outB[(size_t)mrow * NOUT + nc] = (__bf16)g;
                } else {
                    outF[(size_t)nc * BM + mrow] = v;   // NCHW scatter
                }
            }
        }
    }
}

// ---------- instance norm (biased variance, eps 1e-5) ----------
__global__ void inorm_stats(const float* __restrict__ y, float* __restrict__ mean,
                            float* __restrict__ rstd, int N) {
    __shared__ float ss[256], sq[256];
    const int c = blockIdx.x;
    float s = 0.f, q = 0.f;
    for (int m = threadIdx.x; m < BM; m += 256) {
        const float v = y[(size_t)m * N + c];
        s += v; q += v * v;
    }
    ss[threadIdx.x] = s; sq[threadIdx.x] = q;
    __syncthreads();
    for (int st = 128; st > 0; st >>= 1) {
        if (threadIdx.x < st) { ss[threadIdx.x] += ss[threadIdx.x + st]; sq[threadIdx.x] += sq[threadIdx.x + st]; }
        __syncthreads();
    }
    if (threadIdx.x == 0) {
        const float mu  = ss[0] * (1.0f / BM);
        const float var = sq[0] * (1.0f / BM) - mu * mu;
        mean[c] = mu;
        rstd[c] = rsqrtf(var + 1e-5f);
    }
}

__global__ void inorm_apply(const float* __restrict__ y, const float* __restrict__ mean,
                            const float* __restrict__ rstd, __bf16* __restrict__ out) {
    const int i = blockIdx.x * blockDim.x + threadIdx.x;
    const int c = i & 511;
    out[i] = (__bf16)((y[i] - mean[c]) * rstd[c]);
}

extern "C" void kernel_launch(void* const* d_in, const int* in_sizes, int n_in,
                              void* d_out, int out_size, void* d_ws, size_t ws_size,
                              hipStream_t stream) {
    const float* feats = (const float*)d_in[0];   // (1,6,131,16,44)
    const float* l2i   = (const float*)d_in[1];   // (1,6,4,4)
    const float* bev   = (const float*)d_in[2];   // (1,4,128,128,3)
    const float* w1    = (const float*)d_in[3];   // (512,128,3,3)
    const float* w2    = (const float*)d_in[4];   // (512,512,1,1)
    const float* w3    = (const float*)d_in[5];   // (256,512,3,3)
    float* out = (float*)d_out;                   // (1,256,128,128)

    char* ws = (char*)d_ws;
    size_t off = 0;
    auto carve = [&](size_t bytes) -> char* {
        char* p = ws + off;
        off = (off + bytes + 255) & ~(size_t)255;
        return p;
    };
    __bf16* wb1  = (__bf16*)carve(1152ull * 512 * 2);
    __bf16* wb2  = (__bf16*)carve(512ull * 512 * 2);
    __bf16* wb3  = (__bf16*)carve(4608ull * 256 * 2);
    __bf16* x0   = (__bf16*)carve((size_t)BM * 128 * 2);
    float*  y1   = (float*)carve((size_t)BM * 512 * 4);
    __bf16* y1n  = (__bf16*)carve((size_t)BM * 512 * 2);
    __bf16* y2   = (__bf16*)carve((size_t)BM * 512 * 2);
    float*  mean = (float*)carve(512 * 4);
    float*  rstd = (float*)carve(512 * 4);

    // weight repack -> fragment-major bf16
    repack_kernel<<<(512 * 128 * 9 + 255) / 256, 256, 0, stream>>>(w1, wb1, 512, 128, 9);
    repack_kernel<<<(512 * 512 * 1 + 255) / 256, 256, 0, stream>>>(w2, wb2, 512, 512, 1);
    repack_kernel<<<(256 * 512 * 9 + 255) / 256, 256, 0, stream>>>(w3, wb3, 256, 512, 9);

    // sampling + weighting + P-sum -> x0 [16384][128] bf16
    sample_kernel<<<BM, 160, 0, stream>>>(feats, l2i, bev, x0);

    // conv1: 3x3, 128->512, K=1152
    gemm_conv_wmma<128, 7, 9, 512, 0><<<dim3(BM / 64, 512 / 128), 256, 0, stream>>>(
        x0, wb1, y1, nullptr);

    // instance norm -> bf16
    inorm_stats<<<512, 256, 0, stream>>>(y1, mean, rstd, 512);
    inorm_apply<<<(BM * 512) / 256, 256, 0, stream>>>(y1, mean, rstd, y1n);

    // conv2: 1x1, 512->512, GELU -> bf16
    gemm_conv_wmma<512, 9, 1, 512, 1><<<dim3(BM / 64, 512 / 128), 256, 0, stream>>>(
        y1n, wb2, nullptr, y2);

    // conv3: 3x3, 512->256, K=4608, NCHW f32 output
    gemm_conv_wmma<512, 9, 9, 256, 2><<<dim3(BM / 64, 256 / 128), 256, 0, stream>>>(
        y2, wb3, out, nullptr);
}
